// GraphTransformer_42288247996467
// MI455X (gfx1250) — matrix operand order repaired and checked by reading, not compiled, not run on hardware
//
#include <hip/hip_runtime.h>

#ifndef USE_TDM
#define USE_TDM 1
#endif

// ---------------- constants ----------------
constexpr int DIM   = 128;   // feature dim
constexpr int HEADS = 8;
constexpr int LSTR  = 136;   // padded LDS row stride (bf16 elems): 272B = 256B + 16B pad

typedef __attribute__((ext_vector_type(16))) __bf16 v16bf;
typedef __attribute__((ext_vector_type(8)))  __bf16 v8bf;
typedef __attribute__((ext_vector_type(4)))  __bf16 v4bf;
typedef __attribute__((ext_vector_type(8)))  float  v8f;
typedef __attribute__((ext_vector_type(4)))  unsigned int u32x4;
typedef __attribute__((ext_vector_type(8)))  int       i32x8;

static __device__ __forceinline__ __bf16 f2bf(float f) {
  unsigned u = __builtin_bit_cast(unsigned, f);
  unsigned r = u + 0x7FFFu + ((u >> 16) & 1u);           // round-to-nearest-even
  unsigned short h = (unsigned short)(r >> 16);
  return __builtin_bit_cast(__bf16, h);
}

// ------------------------------------------------------------------
// TDM: async load of one 128x128 bf16 tile (row-major, stride 128) into LDS
// with hardware padding producing a 272-byte LDS row stride (LSTR=136 bf16).
//   pad_interval=5 -> pad after every 2^(5+1)=64 DWORDs (=256B = one row)
//   pad_amount  =3 -> insert 4 DWORDs (16B) of padding
// ------------------------------------------------------------------
static __device__ __forceinline__ void tdm_load_tile(const __bf16* gsrc, unsigned lds_off) {
  unsigned long long ga = (unsigned long long)(uintptr_t)gsrc;
  u32x4 g0;
  g0[0] = 1u;                                             // count=1 (valid user D#)
  g0[1] = lds_off;                                        // lds_addr (bytes)
  g0[2] = (unsigned)(ga & 0xFFFFFFFFu);                   // global_addr[31:0]
  g0[3] = (unsigned)((ga >> 32) & 0x1FFFFFFu) | (2u << 30); // global_addr[56:32] | type=2
  i32x8 g1;
  g1[0] = (int)((1u << 16) | (1u << 20) | (5u << 22) | (3u << 25)); // data_size=2B, pad_en, interval, amount
  g1[1] = (int)(128u << 16);                              // tensor_dim0[15:0]=128
  g1[2] = (int)(128u << 16);                              // tensor_dim1[15:0]=128
  g1[3] = (int)(128u << 16);                              // tile_dim0=128
  g1[4] = (int)(128u);                                    // tile_dim1=128
  g1[5] = (int)(128u);                                    // tensor_dim0_stride=128
  g1[6] = 0;
  g1[7] = 0;
  asm volatile("tensor_load_to_lds %0, %1" :: "s"(g0), "s"(g1) : "memory");
}

// ------------------------------------------------------------------
// prep: f32 -> bf16 copy (grid-stride)
// ------------------------------------------------------------------
__global__ __launch_bounds__(256)
void cvt_bf16(const float* __restrict__ src, __bf16* __restrict__ dst, int n)
{
  for (int i = blockIdx.x * 256 + threadIdx.x; i < n; i += gridDim.x * 256)
    dst[i] = f2bf(src[i]);
}

// prep: weight 128x128 f32 (K x N, row-major) -> bf16 transposed (N x K)
__global__ __launch_bounds__(256)
void prep_w(const float* __restrict__ w, __bf16* __restrict__ wt)
{
  int i = blockIdx.x * 256 + threadIdx.x;     // 16384 elems
  int k = i >> 7, n = i & (DIM - 1);
  wt[n * DIM + k] = f2bf(w[i]);
}

// ------------------------------------------------------------------
// Fused (optionally gathered) node GEMM, bf16 inputs, f32 accumulate:
//   out[n,:] = A[ids ? ids[n] : n, :] @ W + bias (+ addend[n,:])
// A: nrows x 128 bf16; Wt: 128x128 bf16 PRE-TRANSPOSED (N-major);
// 256 threads = 8 wave32; each wave -> 16 rows x 128 cols via WMMA bf16.
// ------------------------------------------------------------------
__global__ __launch_bounds__(256)
void gemm_node(const __bf16* __restrict__ A, const int* __restrict__ ids,
               const __bf16* __restrict__ Wsrc, const float* __restrict__ bias,
               const float* __restrict__ addend, float* __restrict__ outF,
               __bf16* __restrict__ outB, int nrows)
{
  __shared__ __align__(16) __bf16 Wt[DIM * LSTR];   // Wt[n*LSTR + k]

  const int tid = threadIdx.x;
#if USE_TDM
  if (tid < 32) {                                    // one wave issues the DMA
    tdm_load_tile(Wsrc, (unsigned)(uintptr_t)(&Wt[0]));
    __builtin_amdgcn_s_wait_tensorcnt(0);
  }
  __syncthreads();
#else
  for (int i = tid; i < (DIM * DIM) / 8; i += 256) {
    int idx = i * 8, n = idx >> 7, k = idx & (DIM - 1);
    *(v8bf*)(Wt + n * LSTR + k) = *(const v8bf*)(Wsrc + n * DIM + k);
  }
  __syncthreads();
#endif

  const int wave = tid >> 5, lane = tid & 31;
  const int rowbase = blockIdx.x * 128 + wave * 16;
  if (rowbase >= nrows) return;                      // wave-uniform

  const int mrow = rowbase + (lane & 15);
  const int arow = ids ? ids[mrow] : mrow;
  const v8bf* __restrict__ ap8 = (const v8bf*)(A + (size_t)arow * DIM);
  const int hsel = lane >> 4;                        // which K-half this lane holds

  v8f acc[8] = {};

  #pragma unroll
  for (int kk = 0; kk < 4; ++kk) {
    // A fragment (16x32 bf16): lane<16 K = s..s+7, s+16..s+23 ; lane>=16 shifted by 8
    const int s1 = kk * 32 + hsel * 8;
    v8bf a0 = ap8[s1 >> 3];
    v8bf a1 = ap8[(s1 + 16) >> 3];
    v16bf afrag;
    #pragma unroll
    for (int j = 0; j < 8; ++j) { afrag[j] = a0[j]; afrag[8 + j] = a1[j]; }

    const int kb = kk * 32 + hsel * 8;               // B (32x16): same K layout as A (lane<16 K=0-7,16-23; lane>=16 shifted by 8)
    #pragma unroll
    for (int t = 0; t < 8; ++t) {
      const int col = t * 16 + (lane & 15);
      const v8bf* bp = (const v8bf*)(Wt + col * LSTR + kb);
      v8bf b0 = bp[0], b1 = bp[2];
      v16bf bfrag;
      #pragma unroll
      for (int j = 0; j < 8; ++j) { bfrag[j] = b0[j]; bfrag[8 + j] = b1[j]; }
      acc[t] = __builtin_amdgcn_wmma_f32_16x16x32_bf16(
          false, afrag, false, bfrag, (short)0, acc[t], false, false);
    }
  }

  // Epilogue: C/D layout -> VGPR r: lane<16 M=r, lane>=16 M=8+r; N = lane&15 in tile
  #pragma unroll
  for (int t = 0; t < 8; ++t) {
    const int col = t * 16 + (lane & 15);
    const float bv = bias[col];
    #pragma unroll
    for (int r = 0; r < 8; ++r) {
      const int row = rowbase + r + hsel * 8;
      float v = acc[t][r] + bv;
      if (addend) v += addend[(size_t)row * DIM + col];
      outF[(size_t)row * DIM + col] = v;
      if (outB) outB[(size_t)row * DIM + col] = f2bf(v);
    }
  }
}

// ------------------------------------------------------------------
// Edge attention logits: one thread per (edge, head)
// ------------------------------------------------------------------
__global__ __launch_bounds__(256)
void edge_att(const int* __restrict__ rows, const int* __restrict__ cols,
              const float* __restrict__ Q, const float* __restrict__ K,
              float* __restrict__ eatt, float* __restrict__ asum, int nedge)
{
  int t = blockIdx.x * 256 + threadIdx.x;
  if (t >= nedge * HEADS) return;
  const int e = t >> 3, h = t & 7;
  const int r = rows[e], c = cols[e];
  const float4* qp = (const float4*)(Q + (size_t)r * DIM + h * 16);
  const float4* kp = (const float4*)(K + (size_t)c * DIM + h * 16);
  float dot = 0.f;
  #pragma unroll
  for (int j = 0; j < 4; ++j) {
    float4 a = qp[j], b = kp[j];
    dot += a.x * b.x + a.y * b.y + a.z * b.z + a.w * b.w;
  }
  dot *= 0.25f;                                        // / sqrt(16)
  dot = fminf(10.f, fmaxf(-10.f, dot));
  const float ex = __expf(dot);
  eatt[t] = ex;
  unsafeAtomicAdd(&asum[r * HEADS + h], ex);
}

// ------------------------------------------------------------------
// Weighted scatter of V: one thread per (edge, head)
// ------------------------------------------------------------------
__global__ __launch_bounds__(256)
void edge_scatter(const int* __restrict__ rows, const int* __restrict__ cols,
                  const float* __restrict__ V, const float* __restrict__ eatt,
                  const float* __restrict__ asum, float* __restrict__ agg, int nedge)
{
  int t = blockIdx.x * 256 + threadIdx.x;
  if (t >= nedge * HEADS) return;
  const int e = t >> 3, h = t & 7;
  const int r = rows[e], c = cols[e];
  const float w = eatt[t] / (asum[r * HEADS + h] + 1e-8f);
  const float4* vp = (const float4*)(V + (size_t)c * DIM + h * 16);
  float* op = agg + (size_t)r * DIM + h * 16;
  #pragma unroll
  for (int j = 0; j < 4; ++j) {
    float4 vv = vp[j];
    unsafeAtomicAdd(op + 4 * j + 0, w * vv.x);
    unsafeAtomicAdd(op + 4 * j + 1, w * vv.y);
    unsafeAtomicAdd(op + 4 * j + 2, w * vv.z);
    unsafeAtomicAdd(op + 4 * j + 3, w * vv.w);
  }
}

// ------------------------------------------------------------------
// Residual + LayerNorm, one wave32 per row; writes f32 X and bf16 mirror XB
// ------------------------------------------------------------------
__global__ __launch_bounds__(256)
void ln_residual(const float* __restrict__ agg, float* __restrict__ x,
                 __bf16* __restrict__ xb,
                 const float* __restrict__ g, const float* __restrict__ b, int n)
{
  const int wave = threadIdx.x >> 5, lane = threadIdx.x & 31;
  const int row = blockIdx.x * 8 + wave;
  if (row >= n) return;
  const float4 av = *(const float4*)(agg + (size_t)row * DIM + lane * 4);
  const float4 xv = *(const float4*)(x + (size_t)row * DIM + lane * 4);
  const float y0 = av.x + xv.x, y1 = av.y + xv.y, y2 = av.z + xv.z, y3 = av.w + xv.w;
  float s  = y0 + y1 + y2 + y3;
  float ss = y0 * y0 + y1 * y1 + y2 * y2 + y3 * y3;
  #pragma unroll
  for (int off = 16; off > 0; off >>= 1) {
    s  += __shfl_xor(s, off, 32);
    ss += __shfl_xor(ss, off, 32);
  }
  const float mu  = s * (1.0f / DIM);
  const float var = ss * (1.0f / DIM) - mu * mu;
  const float inv = rsqrtf(var + 1e-6f);
  const float4 gv = *(const float4*)(g + lane * 4);
  const float4 bv = *(const float4*)(b + lane * 4);
  float4 o;
  o.x = (y0 - mu) * inv * gv.x + bv.x;
  o.y = (y1 - mu) * inv * gv.y + bv.y;
  o.z = (y2 - mu) * inv * gv.z + bv.z;
  o.w = (y3 - mu) * inv * gv.w + bv.w;
  *(float4*)(x + (size_t)row * DIM + lane * 4) = o;
  v4bf ob;
  ob[0] = f2bf(o.x); ob[1] = f2bf(o.y); ob[2] = f2bf(o.z); ob[3] = f2bf(o.w);
  *(v4bf*)(xb + (size_t)row * DIM + lane * 4) = ob;
}

__global__ __launch_bounds__(256)
void zero_kernel(float* __restrict__ p, int n)
{
  int i = blockIdx.x * 256 + threadIdx.x;
  if (i < n) p[i] = 0.f;
}

// ------------------------------------------------------------------
extern "C" void kernel_launch(void* const* d_in, const int* in_sizes, int n_in,
                              void* d_out, int out_size, void* d_ws, size_t ws_size,
                              hipStream_t stream)
{
  const int*   ids  = (const int*)d_in[0];
  const int*   eidx = (const int*)d_in[1];
  const float* emb  = (const float*)d_in[2];
  const float* in_w = (const float*)d_in[3];
  const float* in_b = (const float*)d_in[4];
  const float* pos  = (const float*)d_in[5];
  const float* qw   = (const float*)d_in[6];
  const float* qb   = (const float*)d_in[7];
  const float* kw   = (const float*)d_in[8];
  const float* kb   = (const float*)d_in[9];
  const float* vw   = (const float*)d_in[10];
  const float* vb   = (const float*)d_in[11];
  const float* lng  = (const float*)d_in[12];
  const float* lnb  = (const float*)d_in[13];
  const float* ow   = (const float*)d_in[14];
  const float* ob   = (const float*)d_in[15];
  float* out = (float*)d_out;

  const int N = in_sizes[0];          // 40000 (multiple of 16)
  const int E = in_sizes[1] / 2;      // 640000
  const int nEmb = in_sizes[2];       // N_ENT*128 elems
  const int* rows = eidx;
  const int* cols = eidx + E;

  // f32 workspace: X | Q | K | V | AGG | SUM | EATT   (L2-resident)
  float* X   = (float*)d_ws;
  float* Q   = X   + (size_t)N * DIM;
  float* Kb_ = Q   + (size_t)N * DIM;
  float* V   = Kb_ + (size_t)N * DIM;
  float* AGG = V   + (size_t)N * DIM;
  float* SUM = AGG + (size_t)N * DIM;     // contiguous with AGG -> single zero pass
  float* EAT = SUM + (size_t)N * HEADS;
  // bf16 workspace: XB (activations), EB (embedding mirror), WB (11 transposed weights)
  __bf16* XB = (__bf16*)(EAT + (size_t)E * HEADS);
  __bf16* EB = XB + (size_t)N * DIM;
  __bf16* WB = EB + (size_t)nEmb;

  dim3 blk(256);
  dim3 gemmG((N + 127) / 128);
  dim3 edgeG((E * HEADS + 255) / 256);
  dim3 lnG((N + 7) / 8);
  dim3 zG((N * (DIM + HEADS) + 255) / 256);
  dim3 wG(DIM * DIM / 256);

  // ---- prep: bf16 embedding mirror + 11 pre-transposed bf16 weight tiles ----
  cvt_bf16<<<dim3(2048), blk, 0, stream>>>(emb, EB, nEmb);
  const int WSLOT = DIM * DIM;
  prep_w<<<wG, blk, 0, stream>>>(in_w, WB + 0 * WSLOT);
  for (int i = 0; i < 3; ++i) {
    prep_w<<<wG, blk, 0, stream>>>(qw + (size_t)i * WSLOT, WB + (size_t)(1 + i) * WSLOT);
    prep_w<<<wG, blk, 0, stream>>>(kw + (size_t)i * WSLOT, WB + (size_t)(4 + i) * WSLOT);
    prep_w<<<wG, blk, 0, stream>>>(vw + (size_t)i * WSLOT, WB + (size_t)(7 + i) * WSLOT);
  }
  prep_w<<<wG, blk, 0, stream>>>(ow, WB + (size_t)10 * WSLOT);

  // x = emb[ids] @ in_w + in_b + pos_enc   (writes f32 X and bf16 XB)
  gemm_node<<<gemmG, blk, 0, stream>>>(EB, ids, WB + 0 * WSLOT, in_b, pos, X, XB, N);

  for (int i = 0; i < 3; ++i) {
    zero_kernel<<<zG, blk, 0, stream>>>(AGG, N * (DIM + HEADS));
    gemm_node<<<gemmG, blk, 0, stream>>>(XB, nullptr, WB + (size_t)(1 + i) * WSLOT,
                                         qb + (size_t)i * DIM, nullptr, Q, nullptr, N);
    gemm_node<<<gemmG, blk, 0, stream>>>(XB, nullptr, WB + (size_t)(4 + i) * WSLOT,
                                         kb + (size_t)i * DIM, nullptr, Kb_, nullptr, N);
    gemm_node<<<gemmG, blk, 0, stream>>>(XB, nullptr, WB + (size_t)(7 + i) * WSLOT,
                                         vb + (size_t)i * DIM, nullptr, V, nullptr, N);
    edge_att<<<edgeG, blk, 0, stream>>>(rows, cols, Q, Kb_, EAT, SUM, E);
    edge_scatter<<<edgeG, blk, 0, stream>>>(rows, cols, V, EAT, SUM, AGG, E);
    ln_residual<<<lnG, blk, 0, stream>>>(AGG, X, XB, lng + (size_t)i * DIM,
                                         lnb + (size_t)i * DIM, N);
  }

  // out = x @ out_w + out_b
  gemm_node<<<gemmG, blk, 0, stream>>>(XB, nullptr, WB + (size_t)10 * WSLOT, ob,
                                       nullptr, out, nullptr, N);
}
